// MultiHeadSWCA_54322746360244
// MI455X (gfx1250) — matrix operand order repaired and checked by reading, not compiled, not run on hardware
//
#include <hip/hip_runtime.h>
#include <hip/hip_bf16.h>
#include <stdint.h>

// ---------------------------------------------------------------------------
// Shifted-window multi-head cross attention for MI455X (gfx1250).
// One wave32 per (batch, window, head). BF16 WMMA 16x16x32, f32 accumulate.
// All WMMA operands staged in LDS in fragment-friendly (K-contiguous) layouts
// so every fragment load is a packed ds_load_b64. Softmax is done in
// registers with __shfl_xor row reductions (no f32 LDS scratch).
// ---------------------------------------------------------------------------

typedef __bf16 bf16_t;
typedef bf16_t v16bf __attribute__((ext_vector_type(16)));
typedef float  v8f   __attribute__((ext_vector_type(8)));
typedef uint32_t u32;

#define NB    16      // batch
#define NC    256     // channels
#define NHEAD 8
#define HD    32      // head dim
#define GRD   56      // H == W
#define WSZ   7       // window size
#define DISP  3       // shift
#define NTOK  49      // tokens per window
#define NPAD  64      // padded tokens (4 WMMA row tiles)
#define STRD  36      // bf16 row stride, 32-col tiles (72 B: 8B-aligned, bank-rotating)
#define PSTR  68      // bf16 row stride, 64-col P tile (136 B)
#define VTSTR 68      // bf16 row stride for V^T (32 rows x 64 token cols)

__device__ __forceinline__ u32 pack_bf2(float a, float b) {
  union { bf16_t h[2]; u32 w; } p;
  p.h[0] = (bf16_t)a; p.h[1] = (bf16_t)b;
  return p.w;
}

// ---- fragment loaders (contiguous k order: element j of lane half hi holds k = 16*hi + j) ----------

// A fragment: 16x32 bf16, rows mb.., K cols kb.. from row-major buf.
// Lane L: hi=L>>4, m=L&15; needs dwords {8hi..8hi+7} of its
// row -> four 8-byte LDS loads. Requires stride even, kb % 4 == 0.
__device__ __forceinline__ v16bf frag_a(const bf16_t* buf, int stride, int mb,
                                        int kb, int lane) {
  const int hi = lane >> 4, m = lane & 15;
  const uint2* row = (const uint2*)(buf + (mb + m) * stride + kb);
  union { v16bf v; uint2 d[4]; } u;
  u.d[0] = row[(hi << 2) + 0];
  u.d[1] = row[(hi << 2) + 1];
  u.d[2] = row[(hi << 2) + 2];
  u.d[3] = row[(hi << 2) + 3];
  return u.v;
}

// B fragment with logical transpose: B[k][n] = buf[n][k].
// Lane L: hi=L>>4, n=L&15; element j holds K = kb + hi*16 + j -> 16
// consecutive bf16 from one row -> four 8-byte LDS loads.
__device__ __forceinline__ v16bf frag_bt(const bf16_t* buf, int stride, int kb,
                                         int nb, int lane) {
  const int hi = lane >> 4, n = lane & 15;
  const uint2* row = (const uint2*)(buf + (nb + n) * stride + kb + (hi << 4));
  union { v16bf v; uint2 d[4]; } u;
  u.d[0] = row[0]; u.d[1] = row[1]; u.d[2] = row[2]; u.d[3] = row[3];
  return u.v;
}

// Out(64 x 32) = A(64x32) @ W^T, W natural [d][c]; Out row-major [t][d]
__device__ __forceinline__ void proj_rows(const bf16_t* A, const bf16_t* W,
                                          bf16_t* Out, int lane) {
  const int hi = lane >> 4, n = lane & 15;
  v16bf b0 = frag_bt(W, STRD, 0, 0, lane);
  v16bf b1 = frag_bt(W, STRD, 0, 16, lane);
#pragma unroll
  for (int mt = 0; mt < 4; ++mt) {
    v16bf a = frag_a(A, STRD, mt * 16, 0, lane);
    v8f c0 = {}, c1 = {};
    c0 = __builtin_amdgcn_wmma_f32_16x16x32_bf16(false, a, false, b0, (short)0,
                                                 c0, false, false);
    c1 = __builtin_amdgcn_wmma_f32_16x16x32_bf16(false, a, false, b1, (short)0,
                                                 c1, false, false);
#pragma unroll
    for (int r = 0; r < 8; ++r) {
      int row = mt * 16 + r + (hi << 3);
      Out[row * STRD + n]      = (bf16_t)c0[r];
      Out[row * STRD + 16 + n] = (bf16_t)c1[r];
    }
  }
}

// Same projection but output stored transposed: Out[d][t] (for PV B-side)
__device__ __forceinline__ void proj_cols(const bf16_t* A, const bf16_t* W,
                                          bf16_t* Out, int lane) {
  const int hi = lane >> 4, n = lane & 15;
  v16bf b0 = frag_bt(W, STRD, 0, 0, lane);
  v16bf b1 = frag_bt(W, STRD, 0, 16, lane);
#pragma unroll
  for (int mt = 0; mt < 4; ++mt) {
    v16bf a = frag_a(A, STRD, mt * 16, 0, lane);
    v8f c0 = {}, c1 = {};
    c0 = __builtin_amdgcn_wmma_f32_16x16x32_bf16(false, a, false, b0, (short)0,
                                                 c0, false, false);
    c1 = __builtin_amdgcn_wmma_f32_16x16x32_bf16(false, a, false, b1, (short)0,
                                                 c1, false, false);
#pragma unroll
    for (int r = 0; r < 8; ++r) {
      int row = mt * 16 + r + (hi << 3);               // token index
      Out[(n)      * VTSTR + row] = (bf16_t)c0[r];     // d = n
      Out[(16 + n) * VTSTR + row] = (bf16_t)c1[r];     // d = 16+n
    }
  }
}

__global__ __launch_bounds__(32) void swca_wmma_kernel(
    const float* __restrict__ skip, const float* __restrict__ x,
    const float* __restrict__ Wq, const float* __restrict__ Wk,
    const float* __restrict__ Wv, const float* __restrict__ Wl,
    const float* __restrict__ pe, float* __restrict__ out) {
  const int lane  = threadIdx.x;
  const int win   = blockIdx.x;   // 0..63
  const int head  = blockIdx.y;   // 0..7
  const int batch = blockIdx.z;   // 0..15
  const int wy = win >> 3, wx = win & 7;
  const int hi = lane >> 4, n15 = lane & 15;

  __shared__ __align__(16) bf16_t XA[NPAD * STRD];   // x window, later O
  __shared__ __align__(16) bf16_t SA[NPAD * STRD];   // skip window
  __shared__ __align__(16) bf16_t QB[NPAD * STRD];   // Q row-major
  __shared__ __align__(16) bf16_t KB[NPAD * STRD];   // K row-major
  __shared__ __align__(16) bf16_t VT[HD * VTSTR];    // V transposed [d][t]
  __shared__ __align__(16) bf16_t WQs[HD * STRD];    // weights natural [d][c]
  __shared__ __align__(16) bf16_t WKs[HD * STRD];
  __shared__ __align__(16) bf16_t WVs[HD * STRD];
  __shared__ __align__(16) bf16_t WLs[HD * STRD];
  __shared__ __align__(16) bf16_t PA[NPAD * PSTR];   // softmax probs
  __shared__ float peS[169];
  __shared__ int   pos[NTOK];

  // token -> flattened (rolled) spatial position; shared with output scatter
  for (int t = lane; t < NTOK; t += 32) {
    int ty = t / 7, tx = t % 7;
    int sr = (wy * WSZ + ty + DISP) % GRD;
    int sc = (wx * WSZ + tx + DISP) % GRD;
    pos[t] = sr * GRD + sc;
  }
  for (int e = lane; e < 169; e += 32) peS[e] = pe[e];
  // zero padded token rows 49..63 (keeps Q/K/V pads exactly 0.0)
  {
    u32* xz = (u32*)(XA + NTOK * STRD);
    u32* sz = (u32*)(SA + NTOK * STRD);
    for (int e = lane; e < ((NPAD - NTOK) * STRD) / 2; e += 32) {
      xz[e] = 0u; sz[e] = 0u;
    }
  }
  // weights: natural layout [d][c], pair-packed dword stores
  for (int e = lane; e < 512; e += 32) {        // 512 bf16 pairs per matrix
    int d = e >> 4, c2 = e & 15;
    float2 q = ((const float2*)Wq)[e];
    float2 k = ((const float2*)Wk)[e];
    float2 v = ((const float2*)Wv)[e];
    float2 l = ((const float2*)Wl)[e];
    ((u32*)(WQs + d * STRD))[c2] = pack_bf2(q.x, q.y);
    ((u32*)(WKs + d * STRD))[c2] = pack_bf2(k.x, k.y);
    ((u32*)(WVs + d * STRD))[c2] = pack_bf2(v.x, v.y);
    ((u32*)(WLs + d * STRD))[c2] = pack_bf2(l.x, l.y);
  }

  // gather this head's 32 channels of the window (b128 loads, packed stores)
  const size_t ibase = (size_t)batch * (GRD * GRD) * NC + (size_t)head * HD;
  for (int e = lane; e < NTOK * 8; e += 32) {
    int t = e >> 3, d4 = e & 7;
    size_t gi = ibase + (size_t)pos[t] * NC + d4 * 4;
    float4 xv = *(const float4*)(x + gi);
    float4 sv = *(const float4*)(skip + gi);
    u32* xr = (u32*)(XA + t * STRD + d4 * 4);
    xr[0] = pack_bf2(xv.x, xv.y); xr[1] = pack_bf2(xv.z, xv.w);
    u32* sr = (u32*)(SA + t * STRD + d4 * 4);
    sr[0] = pack_bf2(sv.x, sv.y); sr[1] = pack_bf2(sv.z, sv.w);
  }

  // projections (padded rows stay zero): Q,K row-major; V transposed
  proj_rows(XA, WQs, QB, lane);
  proj_rows(XA, WKs, KB, lane);
  proj_cols(SA, WVs, VT, lane);

  // ---- S = Q K^T * scale + bias + masks; softmax in registers -> PA ------
  const float scale = 0.1767766952966369f;  // 1/sqrt(32)
  const bool mUL = (wy == 7);               // last window row
  const bool mLR = (wx == 7);               // last window col
  // K^T B-fragments: B[k=d][n=tk] = KB[tk][d] (row-contiguous)
  v16bf bk0 = frag_bt(KB, STRD, 0, 0,  lane);
  v16bf bk1 = frag_bt(KB, STRD, 0, 16, lane);
  v16bf bk2 = frag_bt(KB, STRD, 0, 32, lane);
  v16bf bk3 = frag_bt(KB, STRD, 0, 48, lane);
  int kyA[4], kxA[4]; bool kvA[4];
#pragma unroll
  for (int nt = 0; nt < 4; ++nt) {
    int kj = nt * 16 + n15;
    kvA[nt] = (kj < NTOK);
    kyA[nt] = kj / 7; kxA[nt] = kj % 7;
  }
#pragma unroll
  for (int mt = 0; mt < 4; ++mt) {
    v16bf aq = frag_a(QB, STRD, mt * 16, 0, lane);
    v8f s[4];
    s[0] = {}; s[0] = __builtin_amdgcn_wmma_f32_16x16x32_bf16(false, aq, false, bk0, (short)0, s[0], false, false);
    s[1] = {}; s[1] = __builtin_amdgcn_wmma_f32_16x16x32_bf16(false, aq, false, bk1, (short)0, s[1], false, false);
    s[2] = {}; s[2] = __builtin_amdgcn_wmma_f32_16x16x32_bf16(false, aq, false, bk2, (short)0, s[2], false, false);
    s[3] = {}; s[3] = __builtin_amdgcn_wmma_f32_16x16x32_bf16(false, aq, false, bk3, (short)0, s[3], false, false);
#pragma unroll
    for (int r = 0; r < 8; ++r) {
      const int qi = mt * 16 + r + (hi << 3);
      const bool qv = (qi < NTOK);
      const int qy = qi / 7, qx = qi % 7;
      float vals[4];
#pragma unroll
      for (int nt = 0; nt < 4; ++nt) {
        float sv = -1.0e30f;
        if (qv && kvA[nt]) {
          sv = s[nt][r] * scale +
               peS[(kyA[nt] - qy + 6) * 13 + (kxA[nt] - qx + 6)];
          if (mUL && ((qy >= 4) != (kyA[nt] >= 4))) sv = -1.0e30f;
          if (mLR && ((qx >= 4) != (kxA[nt] >= 4))) sv = -1.0e30f;
        }
        vals[nt] = sv;
      }
      // row max / sum live in the 16 lanes of this half (xor bits 0..3)
      float mx = fmaxf(fmaxf(vals[0], vals[1]), fmaxf(vals[2], vals[3]));
      mx = fmaxf(mx, __shfl_xor(mx, 1, 32));
      mx = fmaxf(mx, __shfl_xor(mx, 2, 32));
      mx = fmaxf(mx, __shfl_xor(mx, 4, 32));
      mx = fmaxf(mx, __shfl_xor(mx, 8, 32));
      float sum = 0.0f;
#pragma unroll
      for (int nt = 0; nt < 4; ++nt) {
        float ev = __expf(vals[nt] - mx);
        vals[nt] = ev; sum += ev;
      }
      sum += __shfl_xor(sum, 1, 32);
      sum += __shfl_xor(sum, 2, 32);
      sum += __shfl_xor(sum, 4, 32);
      sum += __shfl_xor(sum, 8, 32);
      const float inv = 1.0f / sum;
#pragma unroll
      for (int nt = 0; nt < 4; ++nt)
        PA[qi * PSTR + nt * 16 + n15] = (bf16_t)(vals[nt] * inv);
    }
  }

  // ---- O = P V (two k-steps); reuse XA for O row-major -------------------
  v16bf bv00 = frag_bt(VT, VTSTR, 0,  0,  lane);
  v16bf bv01 = frag_bt(VT, VTSTR, 0,  16, lane);
  v16bf bv10 = frag_bt(VT, VTSTR, 32, 0,  lane);
  v16bf bv11 = frag_bt(VT, VTSTR, 32, 16, lane);
#pragma unroll
  for (int mt = 0; mt < 4; ++mt) {
    v16bf a0 = frag_a(PA, PSTR, mt * 16, 0,  lane);
    v16bf a1 = frag_a(PA, PSTR, mt * 16, 32, lane);
    v8f c0 = {}, c1 = {};
    c0 = __builtin_amdgcn_wmma_f32_16x16x32_bf16(false, a0, false, bv00, (short)0, c0, false, false);
    c0 = __builtin_amdgcn_wmma_f32_16x16x32_bf16(false, a1, false, bv10, (short)0, c0, false, false);
    c1 = __builtin_amdgcn_wmma_f32_16x16x32_bf16(false, a0, false, bv01, (short)0, c1, false, false);
    c1 = __builtin_amdgcn_wmma_f32_16x16x32_bf16(false, a1, false, bv11, (short)0, c1, false, false);
#pragma unroll
    for (int r = 0; r < 8; ++r) {
      int row = mt * 16 + r + (hi << 3);
      XA[row * STRD + n15]      = (bf16_t)c0[r];
      XA[row * STRD + 16 + n15] = (bf16_t)c1[r];
    }
  }

  // ---- Y = O Wl^T, scatter to global (roll-back == same positions) -------
  v16bf bl0 = frag_bt(WLs, STRD, 0, 0,  lane);
  v16bf bl1 = frag_bt(WLs, STRD, 0, 16, lane);
#pragma unroll
  for (int mt = 0; mt < 4; ++mt) {
    v16bf a = frag_a(XA, STRD, mt * 16, 0, lane);
    v8f c0 = {}, c1 = {};
    c0 = __builtin_amdgcn_wmma_f32_16x16x32_bf16(false, a, false, bl0, (short)0, c0, false, false);
    c1 = __builtin_amdgcn_wmma_f32_16x16x32_bf16(false, a, false, bl1, (short)0, c1, false, false);
#pragma unroll
    for (int r = 0; r < 8; ++r) {
      int row = mt * 16 + r + (hi << 3);
      if (row < NTOK) {
        size_t ob = ibase + (size_t)pos[row] * NC;
        out[ob + n15]      = c0[r];
        out[ob + 16 + n15] = c1[r];
      }
    }
  }
}

extern "C" void kernel_launch(void* const* d_in, const int* in_sizes, int n_in,
                              void* d_out, int out_size, void* d_ws,
                              size_t ws_size, hipStream_t stream) {
  const float* skip = (const float*)d_in[0];
  const float* x    = (const float*)d_in[1];
  const float* Wq   = (const float*)d_in[2];
  const float* Wk   = (const float*)d_in[3];
  const float* Wv   = (const float*)d_in[4];
  const float* Wl   = (const float*)d_in[5];
  const float* pe   = (const float*)d_in[6];
  float* out = (float*)d_out;

  dim3 grid(64, NHEAD, NB);  // (window, head, batch)
  dim3 block(32);            // one wave32
  swca_wmma_kernel<<<grid, block, 0, stream>>>(skip, x, Wq, Wk, Wv, Wl, pe,
                                               out);
}
